// GPT_10548439679026
// MI455X (gfx1250) — matrix operand order repaired and checked by reading, not compiled, not run on hardware
//
#include <hip/hip_runtime.h>
#include <hip/hip_bf16.h>
#include <cmath>

// ---------------- problem constants ----------------
constexpr int Bc  = 4;
constexpr int Tc  = 1024;
constexpr int Dc  = 1024;
constexpr int Hc  = 16;
constexpr int DHc = 64;
constexpr int Lc  = 12;
constexpr int Vm1 = 1024;          // V-1
constexpr int Mr  = Bc * Tc;       // 4096 rows of the activation matrix

typedef __attribute__((ext_vector_type(16))) _Float16 v16h;
typedef __attribute__((ext_vector_type(8)))  _Float16 v8h;
typedef __attribute__((ext_vector_type(8)))  float    v8f;

static __device__ __forceinline__ v16h combine16(v8h a, v8h b) {
  return __builtin_shufflevector(a, b, 0,1,2,3,4,5,6,7,8,9,10,11,12,13,14,15);
}
static __device__ __forceinline__ v8f wmma16(v16h a, v16h b, v8f c) {
  // D = A(16x32 f16) * B(32x16 f16) + C(16x16 f32)
  return __builtin_amdgcn_wmma_f32_16x16x32_f16(false, a, false, b, (short)0, c, false, false);
}
static __device__ __forceinline__ v8f zero8() {
  v8f z;
#pragma unroll
  for (int i = 0; i < 8; ++i) z[i] = 0.0f;
  return z;
}

// ---------------- embedding ----------------
__global__ __launch_bounds__(256) void embed_kernel(const int* __restrict__ idx,
                                                    const float* __restrict__ tok,
                                                    const float* __restrict__ pos,
                                                    float* __restrict__ X) {
  const int bt = blockIdx.x;              // 0..Mr-1
  const int t  = bt & (Tc - 1);
  const int id = idx[bt];
  const float4* te = (const float4*)(tok + (size_t)id * Dc);
  const float4* pe = (const float4*)(pos + (size_t)t  * Dc);
  float4*       xo = (float4*)(X + (size_t)bt * Dc);
  const int i = threadIdx.x;              // Dc/4 == 256
  float4 a = te[i], b = pe[i];
  a.x += b.x; a.y += b.y; a.z += b.z; a.w += b.w;
  xo[i] = a;
}

// ---------------- layernorm (f32 in -> f16 out) ----------------
__global__ __launch_bounds__(256) void ln_kernel(const float* __restrict__ X,
                                                 const float* __restrict__ g,
                                                 const float* __restrict__ bta,
                                                 _Float16* __restrict__ O) {
  __shared__ float red1[8];
  __shared__ float red2[8];
  const int row = blockIdx.x, tid = threadIdx.x;
  const float4 xv = ((const float4*)(X + (size_t)row * Dc))[tid];

  float s = xv.x + xv.y + xv.z + xv.w;
#pragma unroll
  for (int m = 1; m < 32; m <<= 1) s += __shfl_xor(s, m, 32);
  if ((tid & 31) == 0) red1[tid >> 5] = s;
  __syncthreads();
  float mean = 0.f;
#pragma unroll
  for (int i = 0; i < 8; ++i) mean += red1[i];
  mean *= (1.0f / Dc);

  const float dx = xv.x - mean, dy = xv.y - mean, dz = xv.z - mean, dw = xv.w - mean;
  float q = dx*dx + dy*dy + dz*dz + dw*dw;
#pragma unroll
  for (int m = 1; m < 32; m <<= 1) q += __shfl_xor(q, m, 32);
  if ((tid & 31) == 0) red2[tid >> 5] = q;
  __syncthreads();
  float var = 0.f;
#pragma unroll
  for (int i = 0; i < 8; ++i) var += red2[i];
  const float rstd = rsqrtf(var * (1.0f / Dc) + 1e-5f);

  const float4 gv = ((const float4*)g)[tid];
  const float4 bv = ((const float4*)bta)[tid];
  _Float16 h[4];
  h[0] = (_Float16)(dx * rstd * gv.x + bv.x);
  h[1] = (_Float16)(dy * rstd * gv.y + bv.y);
  h[2] = (_Float16)(dz * rstd * gv.z + bv.z);
  h[3] = (_Float16)(dw * rstd * gv.w + bv.w);
  *(uint2*)(O + (size_t)row * Dc + tid * 4) = *(uint2*)h;
}

// ---------------- f32 [K][N] -> f16 transposed [N][K] ----------------
// blockIdx.z selects an independent (in,out) pair via the given z-strides
// (used to fuse the 3 QKV weight transposes into one launch).
__global__ __launch_bounds__(256) void cvt_t_kernel(const float* __restrict__ in,
                                                    _Float16* __restrict__ out,
                                                    int K, int N,
                                                    size_t inStrideZ, size_t outStrideZ) {
  __shared__ float tile[32][33];
  in  += (size_t)blockIdx.z * inStrideZ;
  out += (size_t)blockIdx.z * outStrideZ;
  const int n0 = blockIdx.x * 32, k0 = blockIdx.y * 32;
  const int tx = threadIdx.x & 31, ty = threadIdx.x >> 5;   // ty: 0..7
#pragma unroll
  for (int p = 0; p < 4; ++p)
    tile[ty + p * 8][tx] = in[(size_t)(k0 + ty + p * 8) * N + n0 + tx];
  __syncthreads();
#pragma unroll
  for (int p = 0; p < 4; ++p)
    out[(size_t)(n0 + ty + p * 8) * K + k0 + tx] = (_Float16)tile[tx][ty + p * 8];
}

// ---------------- WMMA GEMM: C[M,N] = A[M,K](f16) * Wt[N,K](f16)^T + bias ----
// EPI: 0 = f32 store, 1 = f32 residual in-place, 2 = GELU -> f16, 3 = QKV scatter
// Software-pipelined: global loads of tile k+1 are issued right after the
// barrier and overlap with the 8 WMMAs of tile k.
template <int EPI>
__global__ __launch_bounds__(256) void gemm_kernel(const _Float16* __restrict__ A,
                                                   const _Float16* __restrict__ Bt,
                                                   const float* __restrict__ bias,
                                                   float* __restrict__ Cf,
                                                   _Float16* __restrict__ Ch,
                                                   _Float16* __restrict__ Qo,
                                                   _Float16* __restrict__ Ko,
                                                   _Float16* __restrict__ Vto,
                                                   int N, int K) {
  constexpr int BM = 128, BN = 128, BK = 32, LST = 40;   // 40-half row pad: no bank conflicts
  __shared__ __align__(16) _Float16 sA[BM * LST];
  __shared__ __align__(16) _Float16 sB[BN * LST];

  const int tid  = threadIdx.x;
  const int lane = tid & 31, wave = tid >> 5;
  const int half = (lane >> 4) & 1, l16 = lane & 15;
  const int wm = wave & 3, wn = wave >> 2;               // 4 waves over M, 2 over N
  const int bm = blockIdx.y * BM, bn = blockIdx.x * BN;

  const _Float16* aP = A  + (size_t)bm * K;
  const _Float16* bP = Bt + (size_t)bn * K;

  // per-thread load slots: 512 chunks of 16B per matrix, 2 per thread
  int lrow[2], lkc[2];
#pragma unroll
  for (int i = 0; i < 2; ++i) {
    const int c = tid + 256 * i;
    lrow[i] = c >> 2;
    lkc[i]  = (c & 3) * 8;
  }

  v8f acc[2][4];
#pragma unroll
  for (int i = 0; i < 2; ++i)
#pragma unroll
    for (int j = 0; j < 4; ++j) acc[i][j] = zero8();

  // prologue: fetch tile 0 into registers
  uint4 ra[2], rb[2];
#pragma unroll
  for (int i = 0; i < 2; ++i) {
    ra[i] = *(const uint4*)&aP[(size_t)lrow[i] * K + lkc[i]];
    rb[i] = *(const uint4*)&bP[(size_t)lrow[i] * K + lkc[i]];
  }

  for (int k0 = 0; k0 < K; k0 += BK) {
    // stage current tile to LDS
#pragma unroll
    for (int i = 0; i < 2; ++i) {
      *(uint4*)&sA[lrow[i] * LST + lkc[i]] = ra[i];
      *(uint4*)&sB[lrow[i] * LST + lkc[i]] = rb[i];
    }
    __syncthreads();

    // issue next tile's global loads (latency hidden behind WMMAs below)
    if (k0 + BK < K) {
#pragma unroll
      for (int i = 0; i < 2; ++i) {
        ra[i] = *(const uint4*)&aP[(size_t)lrow[i] * K + k0 + BK + lkc[i]];
        rb[i] = *(const uint4*)&bP[(size_t)lrow[i] * K + k0 + BK + lkc[i]];
      }
    }

    v16h af[2], bf[4];
#pragma unroll
    for (int mt = 0; mt < 2; ++mt) {
      const int r = wm * 32 + mt * 16 + l16;
      af[mt] = combine16(*(const v8h*)&sA[r * LST + half * 8],
                         *(const v8h*)&sA[r * LST + half * 8 + 16]);
    }
#pragma unroll
    for (int nt = 0; nt < 4; ++nt) {
      const int r = wn * 64 + nt * 16 + l16;
      bf[nt] = combine16(*(const v8h*)&sB[r * LST + half * 8],
                         *(const v8h*)&sB[r * LST + half * 8 + 16]);
    }
#pragma unroll
    for (int mt = 0; mt < 2; ++mt)
#pragma unroll
      for (int nt = 0; nt < 4; ++nt)
        acc[mt][nt] = wmma16(af[mt], bf[nt], acc[mt][nt]);
    __syncthreads();
  }

#pragma unroll
  for (int mt = 0; mt < 2; ++mt) {
#pragma unroll
    for (int nt = 0; nt < 4; ++nt) {
      const int colg = bn + wn * 64 + nt * 16 + l16;
      const float bv = bias ? bias[colg] : 0.0f;
#pragma unroll
      for (int v = 0; v < 8; ++v) {
        const int rowg = bm + wm * 32 + mt * 16 + v + half * 8;
        const float val = acc[mt][nt][v] + bv;
        if constexpr (EPI == 0) {
          Cf[(size_t)rowg * N + colg] = val;
        } else if constexpr (EPI == 1) {
          float* p = &Cf[(size_t)rowg * N + colg];
          *p = *p + val;                                  // residual, in-place
        } else if constexpr (EPI == 2) {
          const float ge = 0.5f * val * (1.0f + erff(val * 0.70710678118654752f));
          Ch[(size_t)rowg * N + colg] = (_Float16)ge;
        } else {                                          // QKV scatter (N == 3*Dc)
          const int c  = colg >> 10;                      // 0:q 1:k 2:v
          const int e  = colg & 1023;
          const int hh = e >> 6, dh = e & 63;
          const int b  = rowg >> 10, t = rowg & 1023;
          const int bh = b * Hc + hh;
          if (c == 0)      Qo [((size_t)bh * Tc  + t ) * DHc + dh] = (_Float16)val;
          else if (c == 1) Ko [((size_t)bh * Tc  + t ) * DHc + dh] = (_Float16)val;
          else             Vto[((size_t)bh * DHc + dh) * Tc  + t ] = (_Float16)val;
        }
      }
    }
  }
}

// ---------------- flash attention (no causal mask, full attention) ----------
// Q,K: [B*H][T][64] f16 ; Vt: [B*H][64][T] f16 ; Y: [B][T][D] f16
__global__ __launch_bounds__(128) void attn_kernel(const _Float16* __restrict__ Q,
                                                   const _Float16* __restrict__ Kt,
                                                   const _Float16* __restrict__ Vt,
                                                   _Float16* __restrict__ Y) {
  __shared__ __align__(16) _Float16 sP[4 * 16 * 40];     // per-wave 16x32 P tile (padded)
  const int tid  = threadIdx.x;
  const int lane = tid & 31, wave = tid >> 5;
  const int half = (lane >> 4) & 1, l16 = lane & 15;
  const int bh = blockIdx.y;
  const int t0 = blockIdx.x * 64 + wave * 16;

  // Q fragments (row = t0 + l16, dh split as two 16x32 A tiles) -- kept in regs
  v16h qa[2];
#pragma unroll
  for (int at = 0; at < 2; ++at) {
    const size_t qb = ((size_t)bh * Tc + t0 + l16) * DHc + at * 32 + half * 8;
    qa[at] = combine16(*(const v8h*)&Q[qb], *(const v8h*)&Q[qb + 16]);
  }

  v8f o[4];
#pragma unroll
  for (int t = 0; t < 4; ++t) o[t] = zero8();
  float mi[8], li[8];
#pragma unroll
  for (int v = 0; v < 8; ++v) { mi[v] = -3.0e38f; li[v] = 0.0f; }

  _Float16* pbase = &sP[wave * 640];
  constexpr float scale = 0.125f;                        // 1/sqrt(64)

  for (int j = 0; j < Tc / 32; ++j) {
    // S = Q * K^T for 32 keys  (two 16x16 tiles)
    v8f s[2];
    s[0] = zero8(); s[1] = zero8();
#pragma unroll
    for (int c = 0; c < 2; ++c) {
#pragma unroll
      for (int ks = 0; ks < 2; ++ks) {
        const size_t ka = ((size_t)bh * Tc + j * 32 + c * 16 + l16) * DHc + ks * 32 + half * 8;
        const v16h kb = combine16(*(const v8h*)&Kt[ka], *(const v8h*)&Kt[ka + 16]);
        s[c] = wmma16(qa[ks], kb, s[c]);
      }
    }
    // online softmax (row lives across one 16-lane half per accumulator slot)
#pragma unroll
    for (int v = 0; v < 8; ++v) {
      const float s0 = s[0][v] * scale, s1 = s[1][v] * scale;
      float r = fmaxf(s0, s1);
#pragma unroll
      for (int m = 1; m < 16; m <<= 1) r = fmaxf(r, __shfl_xor(r, m, 32));
      const float mnew = fmaxf(mi[v], r);
      const float corr = __expf(mi[v] - mnew);
      const float p0 = __expf(s0 - mnew);
      const float p1 = __expf(s1 - mnew);
      float ps = p0 + p1;
#pragma unroll
      for (int m = 1; m < 16; m <<= 1) ps += __shfl_xor(ps, m, 32);
      li[v] = li[v] * corr + ps;
      mi[v] = mnew;
#pragma unroll
      for (int t = 0; t < 4; ++t) o[t][v] *= corr;
      const int prow = v + half * 8;
      pbase[prow * 40 + l16]      = (_Float16)p0;
      pbase[prow * 40 + 16 + l16] = (_Float16)p1;
    }
    __syncthreads();
    // P as A fragment, O += P * V
    const v16h pa = combine16(*(const v8h*)&pbase[l16 * 40 + half * 8],
                              *(const v8h*)&pbase[l16 * 40 + half * 8 + 16]);
#pragma unroll
    for (int t = 0; t < 4; ++t) {
      const size_t va = ((size_t)bh * DHc + t * 16 + l16) * Tc + j * 32 + half * 8;
      const v16h vf = combine16(*(const v8h*)&Vt[va], *(const v8h*)&Vt[va + 16]);
      o[t] = wmma16(pa, vf, o[t]);
    }
    __syncthreads();
  }

  const int b = bh >> 4, hh = bh & 15;
#pragma unroll
  for (int t = 0; t < 4; ++t) {
#pragma unroll
    for (int v = 0; v < 8; ++v) {
      const int row = t0 + v + half * 8;
      const int col = hh * DHc + t * 16 + l16;
      Y[((size_t)b * Tc + row) * Dc + col] = (_Float16)(o[t][v] / li[v]);
    }
  }
}

// ---------------- host orchestration ----------------
extern "C" void kernel_launch(void* const* d_in, const int* in_sizes, int n_in,
                              void* d_out, int out_size, void* d_ws, size_t ws_size,
                              hipStream_t stream) {
  const int*   idx   = (const int*)  d_in[0];
  const float* tok   = (const float*)d_in[1];
  const float* pos   = (const float*)d_in[2];
  const float* ln1g  = (const float*)d_in[3];
  const float* ln1b  = (const float*)d_in[4];
  const float* Wqkv  = (const float*)d_in[5];
  const float* bqkv  = (const float*)d_in[6];
  const float* Wproj = (const float*)d_in[7];
  const float* bproj = (const float*)d_in[8];
  const float* ln2g  = (const float*)d_in[9];
  const float* ln2b  = (const float*)d_in[10];
  const float* W1    = (const float*)d_in[11];
  const float* b1    = (const float*)d_in[12];
  const float* W2    = (const float*)d_in[13];
  const float* b2    = (const float*)d_in[14];
  const float* lnfg  = (const float*)d_in[15];
  const float* lnfb  = (const float*)d_in[16];
  const float* headW = (const float*)d_in[17];
  float* out = (float*)d_out;

  char* w = (char*)d_ws;
  auto alloc = [&](size_t bytes) -> void* {
    void* p = (void*)w;
    w += (bytes + 255) & ~(size_t)255;
    return p;
  };
  float*    x   = (float*)   alloc((size_t)Mr * Dc * 4);            // residual stream (f32)
  _Float16* h16 = (_Float16*)alloc((size_t)Mr * Dc * 2);            // LN output (f16)
  _Float16* q   = (_Float16*)alloc((size_t)Bc * Hc * Tc * DHc * 2);
  _Float16* k   = (_Float16*)alloc((size_t)Bc * Hc * Tc * DHc * 2);
  _Float16* vT  = (_Float16*)alloc((size_t)Bc * Hc * DHc * Tc * 2);
  _Float16* y16 = (_Float16*)alloc((size_t)Mr * Dc * 2);            // attention output
  _Float16* u16 = (_Float16*)alloc((size_t)Mr * 4 * Dc * 2);        // MLP hidden
  _Float16* wt  = (_Float16*)alloc((size_t)4 * Dc * Dc * 2);        // transposed f16 weights

  const dim3 gQKV(3 * Dc / 128, Mr / 128);
  const dim3 gD(Dc / 128, Mr / 128);
  const dim3 gM1(4 * Dc / 128, Mr / 128);

  embed_kernel<<<Mr, 256, 0, stream>>>(idx, tok, pos, x);

  for (int l = 0; l < Lc; ++l) {
    ln_kernel<<<Mr, 256, 0, stream>>>(x, ln1g + (size_t)l * Dc, ln1b + (size_t)l * Dc, h16);
    // fused transpose of the 3 QKV weight matrices (z = c)
    cvt_t_kernel<<<dim3(Dc / 32, Dc / 32, 3), 256, 0, stream>>>(
        Wqkv + (size_t)l * 3 * Dc * Dc, wt, Dc, Dc,
        (size_t)Dc * Dc, (size_t)Dc * Dc);
    gemm_kernel<3><<<gQKV, 256, 0, stream>>>(h16, wt, bqkv + (size_t)l * 3 * Dc,
                                             nullptr, nullptr, q, k, vT, 3 * Dc, Dc);
    attn_kernel<<<dim3(Tc / 64, Bc * Hc), 128, 0, stream>>>(q, k, vT, y16);

    cvt_t_kernel<<<dim3(Dc / 32, Dc / 32, 1), 256, 0, stream>>>(
        Wproj + (size_t)l * Dc * Dc, wt, Dc, Dc, 0, 0);
    gemm_kernel<1><<<gD, 256, 0, stream>>>(y16, wt, bproj + (size_t)l * Dc,
                                           x, nullptr, nullptr, nullptr, nullptr, Dc, Dc);

    ln_kernel<<<Mr, 256, 0, stream>>>(x, ln2g + (size_t)l * Dc, ln2b + (size_t)l * Dc, h16);
    cvt_t_kernel<<<dim3(4 * Dc / 32, Dc / 32, 1), 256, 0, stream>>>(
        W1 + (size_t)l * Dc * 4 * Dc, wt, Dc, 4 * Dc, 0, 0);
    gemm_kernel<2><<<gM1, 256, 0, stream>>>(h16, wt, b1 + (size_t)l * 4 * Dc,
                                            nullptr, u16, nullptr, nullptr, nullptr, 4 * Dc, Dc);
    cvt_t_kernel<<<dim3(Dc / 32, 4 * Dc / 32, 1), 256, 0, stream>>>(
        W2 + (size_t)l * 4 * Dc * Dc, wt, 4 * Dc, Dc, 0, 0);
    gemm_kernel<1><<<gD, 256, 0, stream>>>(u16, wt, b2 + (size_t)l * Dc,
                                           x, nullptr, nullptr, nullptr, nullptr, Dc, 4 * Dc);
  }

  ln_kernel<<<Mr, 256, 0, stream>>>(x, lnfg, lnfb, h16);
  cvt_t_kernel<<<dim3(Vm1 / 32, Dc / 32, 1), 256, 0, stream>>>(headW, wt, Dc, Vm1, 0, 0);
  gemm_kernel<0><<<dim3(Vm1 / 128, Mr / 128), 256, 0, stream>>>(
      h16, wt, nullptr, out, nullptr, nullptr, nullptr, nullptr, Vm1, Dc);
}
